// Network_63556926046752
// MI455X (gfx1250) — matrix operand order repaired and checked by reading, not compiled, not run on hardware
//
#include <hip/hip_runtime.h>
#include <math.h>

typedef __attribute__((ext_vector_type(16))) _Float16 v16h;
typedef __attribute__((ext_vector_type(8)))  _Float16 v8h;
typedef __attribute__((ext_vector_type(8)))  float    v8f;

#define LDA 288            // activation row stride (halfs): 256 feat + 27 dirPE + 5 pad

// ---- workspace layout (in _Float16 elements) ----
#define OFF_L0   0          // [256][64]   xyz_w0^T, K padded 63->64
#define OFF_REST 16384      // 7 x [256][256] xyz_w_rest^T
#define OFF_FEAT 475136     // [256][256]  feat_w^T
#define OFF_DIR  540672     // [128][288]  dir_w^T, K padded 283->288
#define WS_HALFS 577536

// =====================================================================
// Prepass: convert weights to f16, transposed [N][K] with K zero-padding
// =====================================================================
__global__ __launch_bounds__(256) void prep_weights(
    const float* __restrict__ xyz_w0,      // [63][256]
    const float* __restrict__ xyz_w_rest,  // [7][256][256]
    const float* __restrict__ feat_w,      // [256][256]
    const float* __restrict__ dir_w,       // [283][128]
    _Float16* __restrict__ w16)
{
  int i = blockIdx.x * 256 + threadIdx.x;
  if (i >= WS_HALFS) return;
  float v;
  if (i < OFF_REST) {                       // xyz_w0^T [256][64]
    int n = i >> 6, k = i & 63;
    v = (k < 63) ? xyz_w0[k * 256 + n] : 0.0f;
  } else if (i < OFF_FEAT) {                // xyz_w_rest^T
    int j = i - OFF_REST;
    int layer = j >> 16, r = j & 65535;
    int n = r >> 8, k = r & 255;
    v = xyz_w_rest[layer * 65536 + k * 256 + n];
  } else if (i < OFF_DIR) {                 // feat_w^T
    int r = i - OFF_FEAT;
    int n = r >> 8, k = r & 255;
    v = feat_w[k * 256 + n];
  } else {                                  // dir_w^T [128][288]
    int r = i - OFF_DIR;
    int n = r / 288, k = r % 288;
    v = (k < 283) ? dir_w[k * 128 + n] : 0.0f;
  }
  w16[i] = (_Float16)v;
}

// =====================================================================
// Tiled WMMA layer: D[128][N] = act( A[128][K] x W^T[N][K] + bias )
// 8 waves: 4 M-slabs (32 rows) x 2 N-halves. Per wave: 2x NT accum tiles.
// A in LDS row-major [128][LDA] f16; Wt in LDS [N][K] f16.
// =====================================================================
template<int KSTEPS, int NT, bool RELU>
__device__ __forceinline__ void mlp_layer(const _Float16* __restrict__ A,
                                          const _Float16* __restrict__ Wt,
                                          const float* __restrict__ bias,
                                          _Float16* __restrict__ D,
                                          int tid)
{
  constexpr int K = KSTEPS * 32;
  constexpr int NHALF = NT * 16;
  const int lane = tid & 31, wid = tid >> 5;
  const int l16 = lane & 15, lsel = lane >> 4;
  const int rm = (wid >> 1) * 32;        // M slab base row
  const int cn = (wid & 1) * NHALF;      // N half base col

  v8f acc[2][NT];
#pragma unroll
  for (int mt = 0; mt < 2; ++mt)
#pragma unroll
    for (int nt = 0; nt < NT; ++nt)
#pragma unroll
      for (int j = 0; j < 8; ++j) acc[mt][nt][j] = 0.0f;

#pragma unroll
  for (int ks = 0; ks < KSTEPS; ++ks) {
    const int k0 = ks * 32;
    v16h a[2];
#pragma unroll
    for (int mt = 0; mt < 2; ++mt) {
      // 16-bit A 16x32 layout: lanes0-15 K[k0+0..7], lanes16-31 K[k0+8..15]
      // in v0-3; +16 K offset in v4-7.
      const _Float16* ap = A + (rm + mt * 16 + l16) * LDA + k0 + 8 * lsel;
      v8h alo = *(const v8h*)ap;
      v8h ahi = *(const v8h*)(ap + 16);
      a[mt] = __builtin_shufflevector(alo, ahi, 0,1,2,3,4,5,6,7,8,9,10,11,12,13,14,15);
    }
#pragma unroll
    for (int nt = 0; nt < NT; ++nt) {
      // 16-bit B 32x16 layout: lane holds column n, K[k0+8*lsel .. +7] then K[k0+16+8*lsel .. +7]
      const _Float16* wp = Wt + (cn + nt * 16 + l16) * K + k0 + 8 * lsel;
      v8h blo = *(const v8h*)wp;
      v8h bhi = *(const v8h*)(wp + 16);
      v16h b = __builtin_shufflevector(blo, bhi, 0,1,2,3,4,5,6,7,8,9,10,11,12,13,14,15);
      acc[0][nt] = __builtin_amdgcn_wmma_f32_16x16x32_f16(
          false, a[0], false, b, (short)0, acc[0][nt], false, false);
      acc[1][nt] = __builtin_amdgcn_wmma_f32_16x16x32_f16(
          false, a[1], false, b, (short)0, acc[1][nt], false, false);
    }
  }

  // Epilogue: C/D layout: lanes0-15 -> M=r, lanes16-31 -> M=r+8; N = l16 + tilebase
#pragma unroll
  for (int nt = 0; nt < NT; ++nt) {
    const int n = cn + nt * 16 + l16;
    const float bv = bias[n];
#pragma unroll
    for (int mt = 0; mt < 2; ++mt) {
      const int mb = rm + mt * 16 + 8 * lsel;
#pragma unroll
      for (int r = 0; r < 8; ++r) {
        float o = acc[mt][nt][r] + bv;
        if (RELU) o = fmaxf(o, 0.0f);
        D[(mb + r) * LDA + n] = (_Float16)o;
      }
    }
  }
}

__device__ __forceinline__ void copyW(const _Float16* __restrict__ src,
                                      _Float16* __restrict__ dst,
                                      int halfs, int tid)
{
  const uint4* s = (const uint4*)src;
  uint4* d = (uint4*)dst;
  int n = halfs >> 3;   // 8 halfs per uint4
  for (int i = tid; i < n; i += 256) d[i] = s[i];
}

// =====================================================================
// Fused NeRF kernel: one workgroup = 128 points = 2 rays.
// =====================================================================
__global__ __launch_bounds__(256) void nerf_fused(
    const float* __restrict__ ro, const float* __restrict__ rd,
    const float* __restrict__ nearp, const float* __restrict__ farp,
    const _Float16* __restrict__ w16,
    const float* __restrict__ xyz_b0, const float* __restrict__ xyz_b_rest,
    const float* __restrict__ feat_b, const float* __restrict__ alpha_w,
    const float* __restrict__ alpha_b, const float* __restrict__ dir_b,
    const float* __restrict__ rgb_w, const float* __restrict__ rgb_b,
    float* __restrict__ out)
{
  __shared__ _Float16 hA[128 * LDA];     // 73728 B
  __shared__ _Float16 hB[128 * LDA];     // 73728 B
  __shared__ _Float16 wT[256 * 256];     // 131072 B
  __shared__ float s_alpha[128];
  __shared__ float s_rgbv[128 * 3];
  __shared__ float s_dist[128];

  const int tid = threadIdx.x;
  const int wg  = blockIdx.x;

  // ---- Stage P: sample points + positional encodings ----
  if (tid < 128) {
    int g = wg * 128 + tid;
    int ray = g >> 6, s = g & 63;
    float nr = nearp[ray], fr = farp[ray];
    float t = (float)s * (1.0f / 63.0f);
    float z = nr + (fr - nr) * t;
    s_dist[tid] = (fr - nr) * (1.0f / 63.0f);
    float dx = rd[ray * 3 + 0], dy = rd[ray * 3 + 1], dz = rd[ray * 3 + 2];
    float px = ro[ray * 3 + 0] + dx * z;
    float py = ro[ray * 3 + 1] + dy * z;
    float pz = ro[ray * 3 + 2] + dz * z;

    _Float16* a = &hA[tid * LDA];
    a[0] = (_Float16)px; a[1] = (_Float16)py; a[2] = (_Float16)pz;
#pragma unroll
    for (int f = 0; f < 10; ++f) {
      float m = (float)(1 << f);
      a[3 + f * 6 + 0] = (_Float16)__sinf(px * m);
      a[3 + f * 6 + 1] = (_Float16)__sinf(py * m);
      a[3 + f * 6 + 2] = (_Float16)__sinf(pz * m);
      a[3 + f * 6 + 3] = (_Float16)__cosf(px * m);
      a[3 + f * 6 + 4] = (_Float16)__cosf(py * m);
      a[3 + f * 6 + 5] = (_Float16)__cosf(pz * m);
    }
    a[63] = (_Float16)0.0f;   // K pad 63->64

    // dir posenc lives in hB cols [256..287] (xyz layers only write cols 0..255)
    _Float16* b = &hB[tid * LDA + 256];
    b[0] = (_Float16)dx; b[1] = (_Float16)dy; b[2] = (_Float16)dz;
#pragma unroll
    for (int f = 0; f < 4; ++f) {
      float m = (float)(1 << f);
      b[3 + f * 6 + 0] = (_Float16)__sinf(dx * m);
      b[3 + f * 6 + 1] = (_Float16)__sinf(dy * m);
      b[3 + f * 6 + 2] = (_Float16)__sinf(dz * m);
      b[3 + f * 6 + 3] = (_Float16)__cosf(dx * m);
      b[3 + f * 6 + 4] = (_Float16)__cosf(dy * m);
      b[3 + f * 6 + 5] = (_Float16)__cosf(dz * m);
    }
#pragma unroll
    for (int k = 27; k < 32; ++k) b[k] = (_Float16)0.0f;  // pad 283->288
  }

  // ---- Layer 0: [128][64] x [64->256] ----
  copyW(w16 + OFF_L0, wT, 256 * 64, tid);
  __syncthreads();
  mlp_layer<2, 8, true>(hA, wT, xyz_b0, hB, tid);

  // ---- Layers 1..7: 256->256, ping-pong (no LDS-pointer arrays: select
  //      buffers with a ternary to avoid addrspacecast static initializers) ----
  int cur = 1;   // h1 currently in hB
  for (int i = 0; i < 7; ++i) {
    __syncthreads();
    copyW(w16 + OFF_REST + i * 65536, wT, 65536, tid);
    __syncthreads();
    _Float16* src = cur ? hB : hA;
    _Float16* dst = cur ? hA : hB;
    mlp_layer<8, 8, true>(src, wT, xyz_b_rest + i * 256, dst, tid);
    cur ^= 1;
  }
  // h8 now in hA (cur == 0)
  __syncthreads();

  // ---- alpha head: 256 -> 1 (VALU dot) ----
  if (tid < 128) {
    float s = alpha_b[0];
    const _Float16* h = &hA[tid * LDA];
    for (int k = 0; k < 256; ++k) s += (float)h[k] * alpha_w[k];
    s_alpha[tid] = 1.0f - __expf(-fmaxf(s, 0.0f) * s_dist[tid]);
  }

  // ---- feature layer: 256 -> 256 (no relu), dst hB cols 0..255 ----
  __syncthreads();
  copyW(w16 + OFF_FEAT, wT, 65536, tid);
  __syncthreads();
  mlp_layer<8, 8, false>(hA, wT, feat_b, hB, tid);

  // ---- dir layer: [128][288] x [288->128] relu, dst hA cols 0..127 ----
  __syncthreads();
  copyW(w16 + OFF_DIR, wT, 128 * 288, tid);
  __syncthreads();
  mlp_layer<9, 4, true>(hB, wT, dir_b, hA, tid);
  __syncthreads();

  // ---- rgb head: 128 -> 3 (VALU) ----
  if (tid < 128) {
    const _Float16* x = &hA[tid * LDA];
    float r0 = rgb_b[0], r1 = rgb_b[1], r2 = rgb_b[2];
    for (int k = 0; k < 128; ++k) {
      float xv = (float)x[k];
      r0 += xv * rgb_w[k * 3 + 0];
      r1 += xv * rgb_w[k * 3 + 1];
      r2 += xv * rgb_w[k * 3 + 2];
    }
    s_rgbv[tid * 3 + 0] = r0;
    s_rgbv[tid * 3 + 1] = r1;
    s_rgbv[tid * 3 + 2] = r2;
  }
  __syncthreads();

  // ---- volume rendering: 1 thread per ray (2 rays/WG) ----
  if (tid < 2) {
    int ray = wg * 2 + tid;
    float T = 1.0f, accm = 0.0f, c0 = 0.0f, c1 = 0.0f, c2 = 0.0f;
    for (int s = 0; s < 64; ++s) {
      int p = tid * 64 + s;
      float a = s_alpha[p];
      float w = a * T;
      out[12288 + ray * 64 + s] = w;                 // weights [4096][64]
      c0 += w * s_rgbv[p * 3 + 0];
      c1 += w * s_rgbv[p * 3 + 1];
      c2 += w * s_rgbv[p * 3 + 2];
      accm += w;
      T *= (1.0f - a + 1e-10f);
    }
    out[ray * 3 + 0] = c0;                           // rgb_map [4096][3]
    out[ray * 3 + 1] = c1;
    out[ray * 3 + 2] = c2;
    out[274432 + ray] = accm;                        // acc_map [4096]
  }
}

// =====================================================================
extern "C" void kernel_launch(void* const* d_in, const int* in_sizes, int n_in,
                              void* d_out, int out_size, void* d_ws, size_t ws_size,
                              hipStream_t stream)
{
  const float* ro         = (const float*)d_in[0];
  const float* rd         = (const float*)d_in[1];
  const float* nearp      = (const float*)d_in[2];
  const float* farp       = (const float*)d_in[3];
  /* d_in[4] = N_samples (64), hardcoded */
  const float* xyz_w0     = (const float*)d_in[5];
  const float* xyz_b0     = (const float*)d_in[6];
  const float* xyz_w_rest = (const float*)d_in[7];
  const float* xyz_b_rest = (const float*)d_in[8];
  const float* feat_w     = (const float*)d_in[9];
  const float* feat_b     = (const float*)d_in[10];
  const float* alpha_w    = (const float*)d_in[11];
  const float* alpha_b    = (const float*)d_in[12];
  const float* dir_w      = (const float*)d_in[13];
  const float* dir_b      = (const float*)d_in[14];
  const float* rgb_w      = (const float*)d_in[15];
  const float* rgb_b      = (const float*)d_in[16];

  _Float16* w16 = (_Float16*)d_ws;
  float* out = (float*)d_out;

  prep_weights<<<WS_HALFS / 256, 256, 0, stream>>>(xyz_w0, xyz_w_rest, feat_w, dir_w, w16);
  nerf_fused<<<2048, 256, 0, stream>>>(ro, rd, nearp, farp, w16,
                                       xyz_b0, xyz_b_rest, feat_b, alpha_w, alpha_b,
                                       dir_b, rgb_w, rgb_b, out);
}
